// GNNInstanceEncoder_48687749267585
// MI455X (gfx1250) — matrix operand order repaired and checked by reading, not compiled, not run on hardware
//
#include <hip/hip_runtime.h>

// ---------------------------------------------------------------------------
// GNN instance encoder for MI455X (gfx1250, wave32, WMMA).
// All 256x256 matmuls run through v_wmma_f32_16x16x32_f16 (f16 in, f32 acc).
// e-tensor (B,V,V,H) = 100MB f32 stays resident; each layer touches it once
// (fused Ce-GEMM + broadcast-add + sigmoid-gate aggregation + LN + residual).
// ---------------------------------------------------------------------------

#define H   256
#define Vn  160
#define Bn  4
#define BV  (Bn * Vn)          // 640 node rows
#define EROWS (Bn * Vn * Vn)   // 102400 edge rows
#define EPS 1e-5f

typedef __attribute__((ext_vector_type(16))) _Float16 v16h;
typedef __attribute__((ext_vector_type(8)))  float    v8f;

// ---------------- wave32 reduction helpers ---------------------------------
__device__ __forceinline__ void wave_red2(float& a, float& b) {
  #pragma unroll
  for (int off = 16; off >= 1; off >>= 1) {
    a += __shfl_xor(a, off, 32);
    b += __shfl_xor(b, off, 32);
  }
}

// ---------------- WMMA core: 16x16 tile, K=256 ------------------------------
// A fragments in LDS, B fragments in global, both pre-swizzled so each lane's
// 16 halves are contiguous (2x b128 loads per fragment).
// Assumed 16-bit A layout (ISA 7.12.2): lane L row M=L&15; lanes 0-15 hold
// K=0..7,16..23; lanes 16-31 hold K=8..15,24..31.
// Assumed 16-bit B layout: lane L col N=L&15; lanes 0-15 K=0..7,16..23; 16-31 K=8..15,24..31.
__device__ __forceinline__ v8f wmma_k256(const _Float16* aLds,
                                         const _Float16* bFrag, int lane) {
  v8f acc = {0.f, 0.f, 0.f, 0.f, 0.f, 0.f, 0.f, 0.f};
  #pragma unroll
  for (int kt = 0; kt < 8; ++kt) {
    v16h a = *(const v16h*)(aLds  + (size_t)(kt * 32 + lane) * 16);
    v16h b = *(const v16h*)(bFrag + (size_t)(kt * 32 + lane) * 16);
    acc = __builtin_amdgcn_wmma_f32_16x16x32_f16(
        false, a, false, b, (short)0, acc, false, false);
  }
  return acc;
}

// Swizzle-store one f32 element (row m, col k of a 16x256 A tile) into the
// A-fragment LDS layout above.
__device__ __forceinline__ void stage_a_elem(_Float16* aLds, int m, int k, float v) {
  int ktile = k >> 5, kk = k & 31;
  int t  = ((kk >> 4) << 3) | (kk & 7);     // half index within lane
  int ln = m | (((kk >> 3) & 1) << 4);      // lane
  aLds[(ktile * 32 + ln) * 16 + t] = (_Float16)v;
}

// ---------------- weight pre-swizzle (f32 row-major -> B fragments) ---------
__global__ void conv_w(const float* __restrict__ W, _Float16* __restrict__ W16) {
  for (int d = blockIdx.x * 256 + threadIdx.x; d < 65536; d += gridDim.x * 256) {
    int t = d & 15, ln = (d >> 4) & 31, ktile = (d >> 9) & 7, ntile = d >> 12;
    int n = ntile * 16 + (ln & 15);
    int k = ktile * 32 + ((t >> 3) * 16 + (ln >> 4) * 8 + (t & 7));
    W16[d] = (_Float16)W[k * H + n];
  }
}

__global__ void zero_f32(float* p, int n) {
  int i = blockIdx.x * 256 + threadIdx.x;
  if (i < n) p[i] = 0.f;
}

// ---------------- node embedding: h0 = LN(relu(job)+relu(mach)) -------------
__global__ void node_embed(const float* __restrict__ Node,
                           const float* __restrict__ jobW, const float* __restrict__ jobB,
                           const float* __restrict__ machW, const float* __restrict__ machB,
                           const float* __restrict__ g, const float* __restrict__ bb,
                           float* __restrict__ h) {
  int wave = threadIdx.x >> 5, lane = threadIdx.x & 31;
  int row = blockIdx.x * 8 + wave;                 // 0..639
  float x0 = Node[row * 2 + 0], x1 = Node[row * 2 + 1];
  float vals[8]; float s = 0.f, s2 = 0.f;
  #pragma unroll
  for (int r = 0; r < 8; ++r) {
    int k = lane + r * 32;
    float jn = fmaxf(x0 * jobW[k] + jobB[k], 0.f);
    float mn = fmaxf(x1 * machW[k] + machB[k], 0.f);
    float v = jn + mn;
    vals[r] = v; s += v; s2 += v * v;
  }
  wave_red2(s, s2);
  float mu = s * (1.f / H);
  float rstd = rsqrtf(s2 * (1.f / H) - mu * mu + EPS);
  #pragma unroll
  for (int r = 0; r < 8; ++r) {
    int k = lane + r * 32;
    h[row * H + k] = (vals[r] - mu) * rstd * g[k] + bb[k];
  }
}

// ---------------- edge init: e = sine_embed(CG) @ edge_W + edge_b -----------
__global__ void edge_init(const float* __restrict__ CG,
                          const _Float16* __restrict__ edgeW16,
                          const float* __restrict__ edgeB,
                          float* __restrict__ e) {
  __shared__ alignas(32) _Float16 aLds[4096];
  int tid = threadIdx.x, lane = tid & 31, wave = tid >> 5;
  size_t rowBase = (size_t)blockIdx.x * 16;
  const float L2_1E4 = 13.287712379549449f;        // log2(10000)
  for (int idx = tid; idx < 4096; idx += 256) {
    int m = idx >> 8, k = idx & 255;
    float x = CG[rowBase + m];
    float expo = (float)(k & ~1) * (1.f / (float)H);
    float ph = x * exp2f(-expo * L2_1E4);          // x / TEMP^expo
    float v = (k & 1) ? __cosf(ph) : __sinf(ph);
    stage_a_elem(aLds, m, k, v);
  }
  __syncthreads();
  #pragma unroll
  for (int q = 0; q < 2; ++q) {
    int nt = wave * 2 + q;
    v8f acc = wmma_k256(aLds, edgeW16 + nt * 4096, lane);
    int n = nt * 16 + (lane & 15);
    int mOff = 8 * (lane >> 4);
    float bn = edgeB[n];
    #pragma unroll
    for (int r = 0; r < 8; ++r)
      e[(rowBase + mOff + r) * H + n] = acc[r] + bn;
  }
}

// ---------------- generic 256-K GEMM: C = [C +] A @ W16 + bias --------------
__global__ void gemm256(const float* __restrict__ A,
                        const _Float16* __restrict__ W16,
                        const float* __restrict__ bias,
                        float* __restrict__ C, int addC) {
  __shared__ alignas(32) _Float16 aLds[4096];
  int tid = threadIdx.x, lane = tid & 31, wave = tid >> 5;
  const float* arow = A + (size_t)blockIdx.x * 16 * H;
  for (int idx = tid; idx < 4096; idx += 256) {
    int m = idx >> 8, k = idx & 255;
    stage_a_elem(aLds, m, k, arow[m * H + k]);
  }
  __syncthreads();
  #pragma unroll
  for (int q = 0; q < 2; ++q) {
    int nt = wave * 2 + q;
    v8f acc = wmma_k256(aLds, W16 + nt * 4096, lane);
    int n = nt * 16 + (lane & 15);
    int mOff = 8 * (lane >> 4);
    float bn = bias[n];
    #pragma unroll
    for (int r = 0; r < 8; ++r) {
      float* dst = C + (size_t)(blockIdx.x * 16 + mOff + r) * H + n;
      float v = acc[r] + bn;
      if (addC) *dst += v; else *dst = v;
    }
  }
}

// ---------------- fused per-layer edge pass ---------------------------------
// Tile = rows (b,i, j0..j0+15) of e.  Computes:
//   e_new = e@C_W + C_b + Ah[b,j] + Bh[b,i]
//   agg[b,i,:] += sum_j sigmoid(e_new) * Vh[b,j,:]      (atomic)
//   e = e + relu(LN(e_new))                              (in-place, row-local)
__global__ void edge_layer(float* __restrict__ e,
                           const _Float16* __restrict__ CW16,
                           const float* __restrict__ Cb,
                           const float* __restrict__ Ah,
                           const float* __restrict__ Bh,
                           const float* __restrict__ Vh,
                           float* __restrict__ agg,
                           const float* __restrict__ neg,
                           const float* __restrict__ neb) {
  __shared__ alignas(32) _Float16 aLds[4096];
  __shared__ float einLds[4096];
  __shared__ float enewLds[4096];
  int tid = threadIdx.x, lane = tid & 31, wave = tid >> 5;
  int blk = blockIdx.x;                 // b*V*10 blocks
  int jt = blk % 10;
  int bi = blk / 10;                    // b*V + i
  int b  = bi / Vn;
  size_t rowBase = (size_t)bi * Vn + jt * 16;   // flattened edge row (j fast)
  float* erow = e + rowBase * H;

  for (int idx = tid; idx < 4096; idx += 256) {
    int m = idx >> 8, k = idx & 255;
    float v = erow[(size_t)m * H + k];
    einLds[m * H + k] = v;
    stage_a_elem(aLds, m, k, v);
  }
  __syncthreads();

  #pragma unroll
  for (int q = 0; q < 2; ++q) {
    int nt = wave * 2 + q;
    v8f acc = wmma_k256(aLds, CW16 + nt * 4096, lane);
    int n = nt * 16 + (lane & 15);
    int mOff = 8 * (lane >> 4);
    float bTerm = Bh[(size_t)bi * H + n] + Cb[n];   // constant over rows (i)
    float sumAgg = 0.f;
    #pragma unroll
    for (int r = 0; r < 8; ++r) {
      int m = mOff + r;
      int j = jt * 16 + m;
      size_t jrow = (size_t)(b * Vn + j) * H + n;
      float v = acc[r] + bTerm + Ah[jrow];          // Ah varies with j
      enewLds[m * H + n] = v;
      float gate = 1.f / (1.f + __expf(-v));
      sumAgg += gate * Vh[jrow];
    }
    sumAgg += __shfl_xor(sumAgg, 16, 32);           // fold rows 8..15 half
    if (lane < 16) atomicAdd(&agg[(size_t)bi * H + n], sumAgg);
  }
  __syncthreads();

  // LN + relu + residual: each wave owns 2 rows of the 16-row tile
  #pragma unroll
  for (int rr = 0; rr < 2; ++rr) {
    int m = wave * 2 + rr;
    float vals[8]; float s = 0.f, s2 = 0.f;
    #pragma unroll
    for (int r = 0; r < 8; ++r) {
      int n = lane + r * 32;
      float v = enewLds[m * H + n];
      vals[r] = v; s += v; s2 += v * v;
    }
    wave_red2(s, s2);
    float mu = s * (1.f / H);
    float rstd = rsqrtf(s2 * (1.f / H) - mu * mu + EPS);
    #pragma unroll
    for (int r = 0; r < 8; ++r) {
      int n = lane + r * 32;
      float ln = (vals[r] - mu) * rstd * neg[n] + neb[n];
      erow[(size_t)m * H + n] = einLds[m * H + n] + fmaxf(ln, 0.f);
    }
  }
}

// ---------------- node update elementwise chain -----------------------------
// tmp = silu(LN_pl(relu(LN_nh(Uh + agg))))
__global__ void node_update(const float* __restrict__ Uh,
                            const float* __restrict__ agg,
                            const float* __restrict__ nhg, const float* __restrict__ nhb,
                            const float* __restrict__ plg, const float* __restrict__ plb,
                            float* __restrict__ tmp) {
  int wave = threadIdx.x >> 5, lane = threadIdx.x & 31;
  int row = blockIdx.x * 8 + wave;
  float v1[8]; float s = 0.f, s2 = 0.f;
  #pragma unroll
  for (int r = 0; r < 8; ++r) {
    int k = lane + r * 32;
    float x = Uh[(size_t)row * H + k] + agg[(size_t)row * H + k];
    v1[r] = x; s += x; s2 += x * x;
  }
  wave_red2(s, s2);
  float mu = s * (1.f / H);
  float rstd = rsqrtf(s2 * (1.f / H) - mu * mu + EPS);
  float v2[8]; float s3 = 0.f, s4 = 0.f;
  #pragma unroll
  for (int r = 0; r < 8; ++r) {
    int k = lane + r * 32;
    float hn = fmaxf((v1[r] - mu) * rstd * nhg[k] + nhb[k], 0.f);
    v2[r] = hn; s3 += hn; s4 += hn * hn;
  }
  wave_red2(s3, s4);
  float mu2 = s3 * (1.f / H);
  float rstd2 = rsqrtf(s4 * (1.f / H) - mu2 * mu2 + EPS);
  #pragma unroll
  for (int r = 0; r < 8; ++r) {
    int k = lane + r * 32;
    float pl = (v2[r] - mu2) * rstd2 * plg[k] + plb[k];
    tmp[(size_t)row * H + k] = pl / (1.f + __expf(-pl));    // silu
  }
}

// ---------------- tmp = relu(LN(h)) -----------------------------------------
__global__ void ln_relu(const float* __restrict__ h,
                        const float* __restrict__ g, const float* __restrict__ b,
                        float* __restrict__ tmp) {
  int wave = threadIdx.x >> 5, lane = threadIdx.x & 31;
  int row = blockIdx.x * 8 + wave;
  float vals[8]; float s = 0.f, s2 = 0.f;
  #pragma unroll
  for (int r = 0; r < 8; ++r) {
    int k = lane + r * 32;
    float v = h[(size_t)row * H + k];
    vals[r] = v; s += v; s2 += v * v;
  }
  wave_red2(s, s2);
  float mu = s * (1.f / H);
  float rstd = rsqrtf(s2 * (1.f / H) - mu * mu + EPS);
  #pragma unroll
  for (int r = 0; r < 8; ++r) {
    int k = lane + r * 32;
    tmp[(size_t)row * H + k] = fmaxf((vals[r] - mu) * rstd * g[k] + b[k], 0.f);
  }
}

// ---------------------------------------------------------------------------
extern "C" void kernel_launch(void* const* d_in, const int* in_sizes, int n_in,
                              void* d_out, int out_size, void* d_ws, size_t ws_size,
                              hipStream_t stream) {
  (void)in_sizes; (void)n_in; (void)out_size; (void)ws_size;

  // ---- inputs in setup_inputs() order (params flattened in insertion order)
  const float* Node = (const float*)d_in[0];
  const float* CG   = (const float*)d_in[1];
  const float* jobW = (const float*)d_in[2];
  const float* jobB = (const float*)d_in[3];
  const float* machW= (const float*)d_in[4];
  const float* machB= (const float*)d_in[5];
  const float* fuseG= (const float*)d_in[6];
  const float* fuseB= (const float*)d_in[7];
  const float* edgeW= (const float*)d_in[8];
  const float* edgeB= (const float*)d_in[9];
  const float* outG = (const float*)d_in[10];
  const float* outB = (const float*)d_in[11];
  const float* outW = (const float*)d_in[12];
  const float* outB2= (const float*)d_in[13];
  // per-layer params: base 14 + l*18
  auto LP = [&](int l, int k) { return (const float*)d_in[14 + l * 18 + k]; };
  // k: 0 U_W 1 U_b 2 V_W 3 V_b 4 A_W 5 A_b 6 B_W 7 B_b 8 C_W 9 C_b
  //    10 nh_g 11 nh_b 12 ne_g 13 ne_b 14 pl_g 15 pl_b 16 pl_W 17 pl_b2

  // ---- workspace carve-up ----
  char* ws = (char*)d_ws;
  size_t off = 0;
  auto carve = [&](size_t bytes) {
    void* p = ws + off;
    off = (off + bytes + 255) & ~(size_t)255;
    return p;
  };
  float* e    = (float*)carve((size_t)EROWS * H * sizeof(float));  // 100 MB
  float* h    = (float*)carve((size_t)BV * H * sizeof(float));
  float* Uh   = (float*)carve((size_t)BV * H * sizeof(float));
  float* Vh   = (float*)carve((size_t)BV * H * sizeof(float));
  float* Ah   = (float*)carve((size_t)BV * H * sizeof(float));
  float* Bh   = (float*)carve((size_t)BV * H * sizeof(float));
  float* agg  = (float*)carve((size_t)BV * H * sizeof(float));
  float* tmp  = (float*)carve((size_t)BV * H * sizeof(float));
  _Float16* w16 = (_Float16*)carve((size_t)17 * 65536 * sizeof(_Float16));
  auto W16 = [&](int slot) { return w16 + (size_t)slot * 65536; };
  // slots: 0 edge, 1 out, then per layer l: 2+6l {U,V,A,B,C,pl}

  // ---- 1) pre-swizzle all weights to WMMA B fragments (f16) ----
  conv_w<<<64, 256, 0, stream>>>(edgeW, W16(0));
  conv_w<<<64, 256, 0, stream>>>(outW,  W16(1));
  for (int l = 0; l < 3; ++l) {
    conv_w<<<64, 256, 0, stream>>>(LP(l, 0),  W16(2 + 6 * l + 0)); // U
    conv_w<<<64, 256, 0, stream>>>(LP(l, 2),  W16(2 + 6 * l + 1)); // V
    conv_w<<<64, 256, 0, stream>>>(LP(l, 4),  W16(2 + 6 * l + 2)); // A
    conv_w<<<64, 256, 0, stream>>>(LP(l, 6),  W16(2 + 6 * l + 3)); // B
    conv_w<<<64, 256, 0, stream>>>(LP(l, 8),  W16(2 + 6 * l + 4)); // C
    conv_w<<<64, 256, 0, stream>>>(LP(l, 16), W16(2 + 6 * l + 5)); // pl
  }

  // ---- 2) node embedding + 3) edge init ----
  node_embed<<<BV / 8, 256, 0, stream>>>(Node, jobW, jobB, machW, machB,
                                         fuseG, fuseB, h);
  edge_init<<<EROWS / 16, 256, 0, stream>>>(CG, W16(0), edgeB, e);

  // ---- 4) layers ----
  for (int l = 0; l < 3; ++l) {
    int s = 2 + 6 * l;
    zero_f32<<<BV, 256, 0, stream>>>(agg, BV * H);
    gemm256<<<BV / 16, 256, 0, stream>>>(h, W16(s + 0), LP(l, 1), Uh, 0);
    gemm256<<<BV / 16, 256, 0, stream>>>(h, W16(s + 1), LP(l, 3), Vh, 0);
    gemm256<<<BV / 16, 256, 0, stream>>>(h, W16(s + 2), LP(l, 5), Ah, 0);
    gemm256<<<BV / 16, 256, 0, stream>>>(h, W16(s + 3), LP(l, 7), Bh, 0);
    edge_layer<<<EROWS / 16, 256, 0, stream>>>(e, W16(s + 4), LP(l, 9),
                                               Ah, Bh, Vh, agg,
                                               LP(l, 12), LP(l, 13));
    node_update<<<BV / 8, 256, 0, stream>>>(Uh, agg, LP(l, 10), LP(l, 11),
                                            LP(l, 14), LP(l, 15), tmp);
    // h += tmp @ pl_W + pl_b2
    gemm256<<<BV / 16, 256, 0, stream>>>(tmp, W16(s + 5), LP(l, 17), h, 1);
  }

  // ---- 5) output head ----
  ln_relu<<<BV / 8, 256, 0, stream>>>(h, outG, outB, tmp);
  gemm256<<<BV / 16, 256, 0, stream>>>(tmp, W16(1), outB2, (float*)d_out, 0);
}
